// SoftQuantizer_68521908240395
// MI455X (gfx1250) — matrix operand order repaired and checked by reading, not compiled, not run on hardware
//
#include <hip/hip_runtime.h>
#include <hip/hip_bf16.h>

// ---------------------------------------------------------------------------
// SoftQuantizer fused kernel for MI455X (gfx1250, wave32, WMMA)
//   quantized = softmax(-cdist(x,W)) @ W     (fused, logits never hit HBM)
//   indices   = argmax of soft weights
//   ortho     = mean((norm_w @ norm_w^T)^2) - 1/K   (tiny, separate kernel)
// ---------------------------------------------------------------------------

typedef __attribute__((ext_vector_type(16))) __bf16 bf16x16;
typedef __attribute__((ext_vector_type(8)))  float  f32x8;

union BVec { uint4 u[2]; bf16x16 v; };
union Pack8 { __bf16 h[8]; uint4 u; };

__device__ __forceinline__ f32x8 wmma_bf16(bf16x16 a, bf16x16 b, f32x8 c) {
  // D = A(16x32) * B(32x16) + C, f32 accumulate
  return __builtin_amdgcn_wmma_f32_16x16x32_bf16(
      /*neg_a=*/false, a, /*neg_b=*/false, b,
      /*c_mod=*/(short)0, c, /*reuse_a=*/false, /*reuse_b=*/false);
}

#define SQ_N    262144
#define SQ_K    512
#define SQ_D    64
#define SQ_T    512         // threads per block (16 waves)
#define SQ_BM   256         // rows per workgroup (16 waves x 16 rows)

// Dynamic LDS layout (bytes)
#define OFF_WT  0                        // __bf16 Wt[64][512]   (B for GEMM1)
#define OFF_WB  65536                    // __bf16 Wb[512][64]   (B for GEMM2)
#define OFF_W2  131072                   // float  w2[512]
#define OFF_X2  133120                   // float  x2[16][16]
#define OFF_P   134144                   // __bf16 P[16][16][32] staging
#define SQ_LDS  150528

__global__ void __launch_bounds__(SQ_T, 1)
sq_main_kernel(const float* __restrict__ X, const float* __restrict__ W,
               float* __restrict__ quant, float* __restrict__ idxOut)
{
  extern __shared__ __align__(16) unsigned char smem[];
  __bf16* sWt = (__bf16*)(smem + OFF_WT);
  __bf16* sWb = (__bf16*)(smem + OFF_WB);
  float*  sW2 = (float*)(smem + OFF_W2);
  float*  sX2 = (float*)(smem + OFF_X2);
  __bf16* sP  = (__bf16*)(smem + OFF_P);

  const int tid  = threadIdx.x;
  const int wid  = tid >> 5;        // wave id 0..15
  const int lane = tid & 31;
  const int hi   = lane >> 4;       // lane half
  const int ln   = lane & 15;
  const int klo  = hi << 3;         // A-layout k offset for this half

  const int rowBase = blockIdx.x * SQ_BM + wid * 16;
  const float* xr = X + (size_t)(rowBase + ln) * SQ_D;
  // get this wave's X rows moving while we process the codebook
  __builtin_prefetch(xr, 0, 1);
  __builtin_prefetch(xr + 32, 0, 1);

  // ---- codebook -> LDS, single vectorized pass ----
  // Each thread: 8 contiguous floats of one codebook row. Produces
  //   sWb[k][d]   row-major bf16 (GEMM2 B), contiguous b128 store
  //   sWt[d][k]   transposed bf16 (GEMM1 B), 8 scattered b16 stores
  //   sW2[k]      f32 row norm^2 via 3x shfl_xor over the 8-lane row group
  for (int base = tid * 8; base < SQ_K * SQ_D; base += SQ_T * 8) {
    const int k  = base >> 6;
    const int d0 = base & 63;
    const float4 wa = *(const float4*)(W + base);
    const float4 wc = *(const float4*)(W + base + 4);
    float s = 0.f;
    s = fmaf(wa.x, wa.x, s); s = fmaf(wa.y, wa.y, s);
    s = fmaf(wa.z, wa.z, s); s = fmaf(wa.w, wa.w, s);
    s = fmaf(wc.x, wc.x, s); s = fmaf(wc.y, wc.y, s);
    s = fmaf(wc.z, wc.z, s); s = fmaf(wc.w, wc.w, s);
    Pack8 p;
    p.h[0] = (__bf16)wa.x; p.h[1] = (__bf16)wa.y;
    p.h[2] = (__bf16)wa.z; p.h[3] = (__bf16)wa.w;
    p.h[4] = (__bf16)wc.x; p.h[5] = (__bf16)wc.y;
    p.h[6] = (__bf16)wc.z; p.h[7] = (__bf16)wc.w;
    *(uint4*)(sWb + base) = p.u;          // 16B aligned (base % 8 == 0)
#pragma unroll
    for (int e = 0; e < 8; ++e) sWt[(d0 + e) * SQ_K + k] = p.h[e];
    s += __shfl_xor(s, 1, 32);
    s += __shfl_xor(s, 2, 32);
    s += __shfl_xor(s, 4, 32);
    if ((tid & 7) == 0) sW2[k] = s;
  }

  // ---- X tile -> registers in WMMA A-layout (bf16), plus ||x||^2 ----
  float x2p = 0.f;
  bf16x16 A0, A1;     // A = x[16 rows x 64] split into two 16x32 operands
#pragma unroll
  for (int half = 0; half < 2; ++half) {
    bf16x16& A = half ? A1 : A0;
    const int kb = half * 32;
#pragma unroll
    for (int e = 0; e < 8; ++e) {
      const float v = xr[kb + klo + e];
      x2p = fmaf(v, v, x2p);
      A[e] = (__bf16)v;
    }
#pragma unroll
    for (int e = 0; e < 8; ++e) {
      const float v = xr[kb + 16 + klo + e];
      x2p = fmaf(v, v, x2p);
      A[e + 8] = (__bf16)v;
    }
  }
  const float x2full = x2p + __shfl_xor(x2p, 16, 32);  // both halves hold row
  if (hi == 0) sX2[wid * 16 + ln] = x2full;

  __syncthreads();

  float x2v[8];
#pragma unroll
  for (int j = 0; j < 8; ++j) x2v[j] = sX2[wid * 16 + j + 8 * hi];

  f32x8 O[4] = {};                   // 16 rows x 64 cols accumulator
  float lsum[8]; float qmax[8]; int qidx[8];
#pragma unroll
  for (int j = 0; j < 8; ++j) { lsum[j] = 0.f; qmax[j] = -1.f; qidx[j] = 0; }

  __bf16* sPw = sP + wid * 512;      // per-wave 16x32 staging

  // ---- main loop: 16 pairs of 16-codeword tiles (32 codes per pass) ----
  for (int kp = 0; kp < 16; ++kp) {
#pragma unroll
    for (int h = 0; h < 2; ++h) {
      const int c0 = kp * 32 + h * 16;
      // B operands: Wb row c = c0 + ln, d = klo..klo+7 and 16+klo..16+klo+7 (+32)
      BVec B0, B1;
      { const __bf16* p = sWb + (size_t)(c0 + ln) * SQ_D + klo;
        B0.u[0] = *(const uint4*)p; B0.u[1] = *(const uint4*)(p + 16); }
      { const __bf16* p = sWb + (size_t)(c0 + ln) * SQ_D + 32 + klo;
        B1.u[0] = *(const uint4*)p; B1.u[1] = *(const uint4*)(p + 16); }
      f32x8 S = {};
      S = wmma_bf16(A0, B0.v, S);    // d = 0..31
      S = wmma_bf16(A1, B1.v, S);    // d = 32..63
      const float w2v = sW2[c0 + ln];
      const int c = c0 + ln;
#pragma unroll
      for (int j = 0; j < 8; ++j) {
        float d2 = x2v[j] + w2v - 2.0f * S[j];
        d2 = fmaxf(d2, 0.0f);
        const float q = __expf(-__builtin_amdgcn_sqrtf(d2)); // dist <= 0 -> q in (0,1]
        lsum[j] += q;
        if (q > qmax[j]) { qmax[j] = q; qidx[j] = c; }
        sPw[(j + 8 * hi) * 32 + h * 16 + ln] = (__bf16)q;    // C-layout -> LDS
      }
    }
    // reload P (16x32) in A-layout; DS ops are in-order within a wave
    BVec Pa;
    { const __bf16* p = sPw + ln * 32 + klo;
      Pa.u[0] = *(const uint4*)p;            // codes klo..klo+7
      Pa.u[1] = *(const uint4*)(p + 16); }   // codes 16+klo..16+klo+7
    const int cb = kp * 32;
#pragma unroll
    for (int dt = 0; dt < 4; ++dt) {
      BVec B2;                               // Wt row d = dt*16+ln, codes klo.. and 16+klo..
      const __bf16* p = sWt + (size_t)(dt * 16 + ln) * SQ_K + cb + klo;
      B2.u[0] = *(const uint4*)p; B2.u[1] = *(const uint4*)(p + 16);
      O[dt] = wmma_bf16(Pa.v, B2.v, O[dt]);
    }
  }

  // ---- reductions across the 16 lanes sharing each row half ----
#pragma unroll
  for (int j = 0; j < 8; ++j) {
    float s = lsum[j];
#pragma unroll
    for (int m = 1; m < 16; m <<= 1) s += __shfl_xor(s, m, 32);
    lsum[j] = s;
    float bq = qmax[j]; int bi = qidx[j];
#pragma unroll
    for (int m = 1; m < 16; m <<= 1) {
      const float oq = __shfl_xor(bq, m, 32);
      const int   oi = __shfl_xor(bi, m, 32);
      if (oq > bq || (oq == bq && oi < bi)) { bq = oq; bi = oi; }
    }
    qmax[j] = bq; qidx[j] = bi;
  }

  // ---- write quantized (normalize by softmax denominator) + indices ----
#pragma unroll
  for (int j = 0; j < 8; ++j) {
    const float invl = 1.0f / lsum[j];
    const size_t row = (size_t)(rowBase + j + 8 * hi);
#pragma unroll
    for (int dt = 0; dt < 4; ++dt)
      quant[row * SQ_D + dt * 16 + ln] = O[dt][j] * invl;
  }
  if (ln == 0) {
#pragma unroll
    for (int j = 0; j < 8; ++j)
      idxOut[rowBase + 8 * hi + j] = (float)qidx[j];
  }
}

// ---------------------------------------------------------------------------
// Orthogonality loss: mean((norm_w @ norm_w^T)^2) - 1/K. 33 MFLOP -> 1 block.
// ---------------------------------------------------------------------------
#define ORTHO_LDS (SQ_K * SQ_D * 4)

__global__ void __launch_bounds__(256, 1)
sq_ortho_kernel(const float* __restrict__ W, float* __restrict__ lossOut)
{
  extern __shared__ __align__(16) unsigned char smem[];
  float* sNW = (float*)smem;               // [512][64] normalized rows
  __shared__ float warpSums[8];
  const int tid = threadIdx.x;

  for (int k = tid; k < SQ_K; k += 256) {
    const float* wr = W + k * SQ_D;
    float s = 0.f;
#pragma unroll
    for (int d = 0; d < SQ_D; ++d) { const float v = wr[d]; s = fmaf(v, v, s); }
    const float inv = 1.0f / fmaxf(__builtin_amdgcn_sqrtf(s), 1e-12f);
#pragma unroll
    for (int d = 0; d < SQ_D; ++d) sNW[k * SQ_D + d] = wr[d] * inv;
  }
  __syncthreads();

  float acc = 0.f;
  for (int i = tid; i < SQ_K; i += 256) {
    float ri[SQ_D];
#pragma unroll
    for (int d = 0; d < SQ_D; ++d) ri[d] = sNW[i * SQ_D + d];
    for (int j = 0; j < SQ_K; ++j) {
      float dot = 0.f;
#pragma unroll
      for (int d = 0; d < SQ_D; ++d) dot = fmaf(ri[d], sNW[j * SQ_D + d], dot);
      acc = fmaf(dot, dot, acc);
    }
  }
#pragma unroll
  for (int m = 1; m < 32; m <<= 1) acc += __shfl_xor(acc, m, 32);
  if ((tid & 31) == 0) warpSums[tid >> 5] = acc;
  __syncthreads();
  if (tid == 0) {
    float t = 0.f;
    for (int w = 0; w < 8; ++w) t += warpSums[w];
    lossOut[0] = t / (float)(SQ_K * SQ_K) - 1.0f / (float)SQ_K;
  }
}

// ---------------------------------------------------------------------------
extern "C" void kernel_launch(void* const* d_in, const int* in_sizes, int n_in,
                              void* d_out, int out_size, void* d_ws, size_t ws_size,
                              hipStream_t stream) {
  (void)in_sizes; (void)n_in; (void)d_ws; (void)ws_size; (void)out_size;
  const float* X = (const float*)d_in[0];   // [N, D] f32
  const float* W = (const float*)d_in[1];   // [K, D] f32
  float* out   = (float*)d_out;
  float* quant = out;                                   // N*D
  float* loss  = out + (size_t)SQ_N * SQ_D;             // 1
  float* idx   = loss + 1;                              // N (as float)

  // allow >64KB dynamic LDS (idempotent, deterministic per call)
  hipFuncSetAttribute(reinterpret_cast<const void*>(sq_main_kernel),
                      hipFuncAttributeMaxDynamicSharedMemorySize, SQ_LDS);
  hipFuncSetAttribute(reinterpret_cast<const void*>(sq_ortho_kernel),
                      hipFuncAttributeMaxDynamicSharedMemorySize, ORTHO_LDS);

  sq_main_kernel<<<SQ_N / SQ_BM, SQ_T, SQ_LDS, stream>>>(X, W, quant, idx);
  sq_ortho_kernel<<<1, 256, ORTHO_LDS, stream>>>(W, loss);
}
